// PatchLSTM_87840671138303
// MI455X (gfx1250) — matrix operand order repaired and checked by reading, not compiled, not run on hardware
//
#include <hip/hip_runtime.h>
#include <math.h>

// ---------------------------------------------------------------------------
// PatchLSTM forward for MI455X (gfx1250, wave32, WMMA).
// All heavy GEMMs use v_wmma_f32_16x16x32_bf16 (bf16 inputs, f32 accumulate).
// Weights are transposed once to [N][K] bf16 so every WMMA fragment load is a
// contiguous 16B (uint4) load in the same contiguous k order for A and B.
// K-loops are kept at unroll(1): the live set per iteration stays < ~100
// VGPRs, avoiding the scratch spills seen with full unrolling; latency is
// hidden by 8 waves/workgroup plus explicit global_prefetch of the next chunk.
// ---------------------------------------------------------------------------

typedef __attribute__((ext_vector_type(16))) __bf16 v16bf;
typedef __attribute__((ext_vector_type(8)))  float  v8f;

union FragB16 {
  v16bf v;
  uint4 q[2];
};

__device__ __forceinline__ unsigned short f2bf(float x) {
  unsigned int u = __float_as_uint(x);
  u += 0x7FFFu + ((u >> 16) & 1u);      // round-to-nearest-even
  return (unsigned short)(u >> 16);
}
__device__ __forceinline__ float bf2f(unsigned short b) {
  return __uint_as_float(((unsigned int)b) << 16);
}
__device__ __forceinline__ float sigmoidf(float x) {
  return 1.0f / (1.0f + __expf(-x));
}

__device__ __forceinline__ v8f wmma_bf16(v16bf a, v16bf b, v8f c) {
  return __builtin_amdgcn_wmma_f32_16x16x32_bf16(false, a, false, b, (short)0, c,
                                                 false, false);
}

// A fragment: 16x32 bf16, row-major source. base points at (row0, k=0).
// lane&15 selects row; lane<16 holds K=kb..kb+15,
// lane>=16 holds K=kb+16..kb+31 (contiguous, matching B).
__device__ __forceinline__ v16bf load_frag_a(const unsigned short* base, int lda,
                                             int kb, int lane) {
  int hi = lane >> 4, ln = lane & 15;
  const unsigned short* p = base + (size_t)ln * lda + kb + hi * 16;
  FragB16 f;
  f.q[0] = *reinterpret_cast<const uint4*>(p);
  f.q[1] = *reinterpret_cast<const uint4*>(p + 8);
  return f.v;
}

// B fragment: 32x16 bf16 from a transposed [N][K] weight. base points at
// (col0 row of Bt, k=0). lane&15 selects N column; lane<16 holds K=kb..kb+15,
// lane>=16 holds K=kb+16..kb+31 (contiguous in Bt).
__device__ __forceinline__ v16bf load_frag_b(const unsigned short* base, int ldb,
                                             int kb, int lane) {
  int hi = lane >> 4, ln = lane & 15;
  const unsigned short* p = base + (size_t)ln * ldb + kb + hi * 16;
  FragB16 f;
  f.q[0] = *reinterpret_cast<const uint4*>(p);
  f.q[1] = *reinterpret_cast<const uint4*>(p + 8);
  return f.v;
}

// ---------------------------------------------------------------------------
// RevIN statistics: one block per series (1024 series x 512 samples).
// ---------------------------------------------------------------------------
__global__ void __launch_bounds__(256) revin_stats_kernel(
    const float* __restrict__ series, float* __restrict__ mean,
    float* __restrict__ stdv, float* __restrict__ invstd) {
  int s = blockIdx.x, tid = threadIdx.x;
  __shared__ float r1[256], r2[256];
  float v0 = series[(size_t)s * 512 + tid];
  float v1 = series[(size_t)s * 512 + 256 + tid];
  r1[tid] = v0 + v1;
  r2[tid] = v0 * v0 + v1 * v1;
  __syncthreads();
  for (int st = 128; st > 0; st >>= 1) {
    if (tid < st) { r1[tid] += r1[tid + st]; r2[tid] += r2[tid + st]; }
    __syncthreads();
  }
  if (tid == 0) {
    float mu  = r1[0] * (1.0f / 512.0f);
    float var = r2[0] * (1.0f / 512.0f) - mu * mu;
    float sd  = sqrtf(var + 1e-5f);
    mean[s] = mu; stdv[s] = sd; invstd[s] = 1.0f / sd;
  }
}

// ---------------------------------------------------------------------------
// Patch extraction + up-projection: one block per (series, patch).
// X[s][t][h] = sum_p normed(s, t*8+p) * up_w[p][h], stored bf16.
// ---------------------------------------------------------------------------
__global__ void __launch_bounds__(256) patch_up_kernel(
    const float* __restrict__ series, const float* __restrict__ mean,
    const float* __restrict__ invstd, const float* __restrict__ rs,
    const float* __restrict__ rb, const float* __restrict__ up_w,
    unsigned short* __restrict__ X) {
  int bid = blockIdx.x;           // s*63 + t
  int s = bid / 63, t = bid % 63;
  int m = s & 7;
  __shared__ float pv[16];
  int tid = threadIdx.x;
  if (tid < 16) {
    float v = series[(size_t)s * 512 + t * 8 + tid];
    pv[tid] = (v - mean[s]) * invstd[s] * rs[m] + rb[m];
  }
  __syncthreads();
  float acc = 0.0f;
#pragma unroll
  for (int p = 0; p < 16; p++) acc += pv[p] * up_w[p * 256 + tid];
  X[(size_t)bid * 256 + tid] = f2bf(acc);
}

// ---------------------------------------------------------------------------
// Transpose + bf16 conversion: dst[n][colOff + k] = src[k][n].
// ---------------------------------------------------------------------------
__global__ void __launch_bounds__(256) transpose_bf16_kernel(
    const float* __restrict__ src, int K, int N, unsigned short* __restrict__ dst,
    int dstStride, int colOff) {
  int idx = blockIdx.x * 256 + threadIdx.x;
  if (idx >= K * N) return;
  int k = idx / N, n = idx % N;
  dst[(size_t)n * dstStride + colOff + k] = f2bf(src[idx]);
}

// ---------------------------------------------------------------------------
// Fused LSTM step: z = [x_t | h_{t-1}] @ W^T (K=768) + b ; gate nonlinearity;
// c/h update; h written to ping-pong buffer and to attention-layout HS.
// Grid (16, 8): 64-row x 64-h-col tiles; each tile owns its 4 gate tiles.
// ---------------------------------------------------------------------------
__global__ void __launch_bounds__(256) lstm_step_kernel(
    const unsigned short* __restrict__ X,      // [1024][63][256] bf16
    const unsigned short* __restrict__ Hprev,  // [1024][512] bf16
    unsigned short* __restrict__ Hnext,        // [1024][512] bf16
    float* __restrict__ C,                     // [1024][512] f32
    const unsigned short* __restrict__ Wt,     // [2048][768] bf16 (transposed)
    const float* __restrict__ bias,            // [2048]
    unsigned short* __restrict__ HS,           // [64512][512] bf16, attn layout
    int t) {
  int lane = threadIdx.x & 31, wave = threadIdx.x >> 5;
  int hi = lane >> 4, ln = lane & 15;
  int rowTile = blockIdx.x * 64, colTile = blockIdx.y * 64;
  for (int sub = wave; sub < 16; sub += 8) {
    int row0 = rowTile + (sub >> 2) * 16;
    int col0 = colTile + (sub & 3) * 16;
    v8f acc[4] = {};
    const unsigned short* xbase = X + ((size_t)row0 * 63 + t) * 256;
    const unsigned short* hbase = Hprev + (size_t)row0 * 512;
    // Per-lane B row pointers, one per gate (weights are [N=2048][K=768]).
    const unsigned short* b0 = Wt + (size_t)(0 * 512 + col0 + ln) * 768;
    const unsigned short* b1 = Wt + (size_t)(1 * 512 + col0 + ln) * 768;
    const unsigned short* b2 = Wt + (size_t)(2 * 512 + col0 + ln) * 768;
    const unsigned short* b3 = Wt + (size_t)(3 * 512 + col0 + ln) * 768;
    int bkoff = hi * 16;
#pragma unroll 1
    for (int kb = 0; kb < 768; kb += 32) {
      v16bf a = (kb < 256) ? load_frag_a(xbase, 63 * 256, kb, lane)
                           : load_frag_a(hbase, 512, kb - 256, lane);
      FragB16 fb;
      // gate 0
      fb.q[0] = *reinterpret_cast<const uint4*>(b0 + kb + bkoff);
      fb.q[1] = *reinterpret_cast<const uint4*>(b0 + kb + bkoff + 8);
      acc[0] = wmma_bf16(a, fb.v, acc[0]);
      // gate 1
      fb.q[0] = *reinterpret_cast<const uint4*>(b1 + kb + bkoff);
      fb.q[1] = *reinterpret_cast<const uint4*>(b1 + kb + bkoff + 8);
      acc[1] = wmma_bf16(a, fb.v, acc[1]);
      // gate 2
      fb.q[0] = *reinterpret_cast<const uint4*>(b2 + kb + bkoff);
      fb.q[1] = *reinterpret_cast<const uint4*>(b2 + kb + bkoff + 8);
      acc[2] = wmma_bf16(a, fb.v, acc[2]);
      // gate 3
      fb.q[0] = *reinterpret_cast<const uint4*>(b3 + kb + bkoff);
      fb.q[1] = *reinterpret_cast<const uint4*>(b3 + kb + bkoff + 8);
      acc[3] = wmma_bf16(a, fb.v, acc[3]);
      // Walk ahead of the loop in L2 (global_prefetch_b8).
      if (kb + 32 < 768) {
        __builtin_prefetch(b0 + kb + 32 + bkoff, 0, 1);
        __builtin_prefetch(b2 + kb + 32 + bkoff, 0, 1);
      }
    }
    int col = col0 + ln;
#pragma unroll
    for (int r = 0; r < 8; r++) {
      int row = row0 + r + 8 * hi;
      float iv = acc[0][r] + bias[col];
      float fv = acc[1][r] + bias[512 + col];
      float gv = acc[2][r] + bias[1024 + col];
      float ov = acc[3][r] + bias[1536 + col];
      size_t ci = (size_t)row * 512 + col;
      float cn = sigmoidf(fv) * C[ci] + sigmoidf(iv) * tanhf(gv);
      C[ci] = cn;
      float h = sigmoidf(ov) * tanhf(cn);
      unsigned short hb = f2bf(h);
      Hnext[ci] = hb;
      // series s=row: b=s>>7, d=(s>>3)&15, m=s&7 -> attn row ((b*8+m)*63+t)*16+d
      int bb = row >> 7, dd = (row >> 3) & 15, mm = row & 7;
      size_t ar = ((size_t)(bb * 8 + mm) * 63 + t) * 16 + dd;
      HS[ar * 512 + col] = hb;
    }
  }
}

// ---------------------------------------------------------------------------
// Generic WMMA GEMM: C[M x N] = A(bf16,[M][lda]) @ Bt(bf16,[N][ldb])^T,
// f32 accumulate, output f32 or bf16. M multiple of 64; N edge masked.
// ---------------------------------------------------------------------------
__global__ void __launch_bounds__(256) gemm_bf16_kernel(
    const unsigned short* __restrict__ A, int lda,
    const unsigned short* __restrict__ Bt, int ldb, int N, int K,
    float* __restrict__ outF, unsigned short* __restrict__ outB, int ldo) {
  int lane = threadIdx.x & 31, wave = threadIdx.x >> 5;
  int hi = lane >> 4, ln = lane & 15;
  int tileM = blockIdx.x * 64, tileN = blockIdx.y * 64;
  for (int sub = wave; sub < 16; sub += 8) {
    int row0 = tileM + (sub >> 2) * 16;
    int col0 = tileN + (sub & 3) * 16;
    const unsigned short* abase = A + (size_t)row0 * lda;
    int n = col0 + ln;
    bool bvalid = (n < N);
    const unsigned short* bbase = Bt + (size_t)(bvalid ? n : 0) * ldb + hi * 16;
    v8f acc = {};
#pragma unroll 1
    for (int kb = 0; kb < K; kb += 32) {
      v16bf a = load_frag_a(abase, lda, kb, lane);
      FragB16 fb;
      if (bvalid) {
        fb.q[0] = *reinterpret_cast<const uint4*>(bbase + kb);
        fb.q[1] = *reinterpret_cast<const uint4*>(bbase + kb + 8);
      } else {
        fb.q[0] = make_uint4(0u, 0u, 0u, 0u);
        fb.q[1] = make_uint4(0u, 0u, 0u, 0u);
      }
      acc = wmma_bf16(a, fb.v, acc);
      if (kb + 32 < K) __builtin_prefetch(bbase + kb + 32, 0, 1);
    }
    if (bvalid) {
      if (outF) {
#pragma unroll
        for (int r = 0; r < 8; r++)
          outF[(size_t)(row0 + r + 8 * hi) * ldo + n] = acc[r];
      } else {
#pragma unroll
        for (int r = 0; r < 8; r++)
          outB[(size_t)(row0 + r + 8 * hi) * ldo + n] = f2bf(acc[r]);
      }
    }
  }
}

// ---------------------------------------------------------------------------
// RMS norm over W elements per row (bf16 in -> bf16 out).
// ---------------------------------------------------------------------------
__global__ void __launch_bounds__(256) rmsnorm_kernel(
    const unsigned short* __restrict__ in, int rowStride, int W,
    const float* __restrict__ scale, unsigned short* __restrict__ out,
    int outStride) {
  int row = blockIdx.x, tid = threadIdx.x;
  __shared__ float red[256];
  float ss = 0.0f;
  for (int i = tid; i < W; i += 256) {
    float v = bf2f(in[(size_t)row * rowStride + i]);
    ss += v * v;
  }
  red[tid] = ss;
  __syncthreads();
  for (int st = 128; st > 0; st >>= 1) {
    if (tid < st) red[tid] += red[tid + st];
    __syncthreads();
  }
  float inv = rsqrtf(red[0] / (float)W + 1e-6f);
  for (int i = tid; i < W; i += 256) {
    float v = bf2f(in[(size_t)row * rowStride + i]);
    out[(size_t)row * outStride + i] = f2bf(v * inv * scale[i]);
  }
}

// ---------------------------------------------------------------------------
// GQA attention core: one block per (group of 16 tokens, head). 8 heads,
// hd=64, 2 KV heads (rep=4). Tiny per-group work -> LDS/VALU.
// ---------------------------------------------------------------------------
__global__ void __launch_bounds__(256) attn_kernel(
    const unsigned short* __restrict__ Qb,  // [64512][512]
    const unsigned short* __restrict__ Kb,  // [64512][128]
    const unsigned short* __restrict__ Vb,  // [64512][128]
    unsigned short* __restrict__ AO) {      // [64512][512]
  int g = blockIdx.x >> 3, h = blockIdx.x & 7, kvh = h >> 2;
  __shared__ float q[16][64], kk[16][64], vv[16][64], sc[16][16];
  int tid = threadIdx.x;
  for (int i = tid; i < 1024; i += 256) {
    int tok = i >> 6, e = i & 63;
    size_t rr = (size_t)(g * 16 + tok);
    q[tok][e]  = bf2f(Qb[rr * 512 + h * 64 + e]);
    kk[tok][e] = bf2f(Kb[rr * 128 + kvh * 64 + e]);
    vv[tok][e] = bf2f(Vb[rr * 128 + kvh * 64 + e]);
  }
  __syncthreads();
  int qi = tid >> 4, ki = tid & 15;
  float s = 0.0f;
#pragma unroll 8
  for (int e = 0; e < 64; e++) s += q[qi][e] * kk[ki][e];
  sc[qi][ki] = s * 0.125f;   // 1/sqrt(64)
  __syncthreads();
  if (tid < 16) {
    float mx = -1e30f;
    for (int k2 = 0; k2 < 16; k2++) mx = fmaxf(mx, sc[tid][k2]);
    float sm = 0.0f;
    for (int k2 = 0; k2 < 16; k2++) {
      float e2 = __expf(sc[tid][k2] - mx);
      sc[tid][k2] = e2; sm += e2;
    }
    float inv = 1.0f / sm;
    for (int k2 = 0; k2 < 16; k2++) sc[tid][k2] *= inv;
  }
  __syncthreads();
  for (int i = tid; i < 1024; i += 256) {
    int tok = i >> 6, e = i & 63;
    float a = 0.0f;
#pragma unroll
    for (int k2 = 0; k2 < 16; k2++) a += sc[tok][k2] * vv[k2][e];
    AO[((size_t)(g * 16 + tok)) * 512 + h * 64 + e] = f2bf(a);
  }
}

// ---------------------------------------------------------------------------
// Permute bel [(b,m,t,d)][256] -> X [(b,d,m)][t][256].
// ---------------------------------------------------------------------------
__global__ void __launch_bounds__(256) permute_kernel(
    const unsigned short* __restrict__ bel, unsigned short* __restrict__ X) {
  size_t i = (size_t)blockIdx.x * 256 + threadIdx.x;
  int r = (int)(i >> 8), c = (int)(i & 255);
  int d = r & 15;
  int t = (r >> 4) % 63;
  int bm = r / (16 * 63);
  int b = bm >> 3, m = bm & 7;
  int s = (b * 16 + d) * 8 + m;
  X[((size_t)s * 63 + t) * 256 + c] = bel[i];
}

// ---------------------------------------------------------------------------
// Final head: denormalize point/quant and write output (4, b, d, m, 96).
// ---------------------------------------------------------------------------
__global__ void __launch_bounds__(256) head_final_kernel(
    const float* __restrict__ POINT,  // [1024][96]
    const float* __restrict__ QUANT,  // [1024][288]
    const float* __restrict__ rs, const float* __restrict__ rb,
    const float* __restrict__ mean, const float* __restrict__ stdv,
    float* __restrict__ out) {
  int idx = blockIdx.x * 256 + threadIdx.x;   // 4*1024*96
  int h = idx % 96;
  int s = (idx / 96) & 1023;
  int qsl = idx / (96 * 1024);
  float y = (qsl == 0) ? POINT[s * 96 + h] : QUANT[s * 288 + h * 3 + (qsl - 1)];
  int m = s & 7;
  out[idx] = (y - rb[m]) / rs[m] * stdv[s] + mean[s];
}

// ---------------------------------------------------------------------------
extern "C" void kernel_launch(void* const* d_in, const int* in_sizes, int n_in,
                              void* d_out, int out_size, void* d_ws, size_t ws_size,
                              hipStream_t stream) {
  (void)in_sizes; (void)n_in; (void)out_size; (void)ws_size;
  const float* series     = (const float*)d_in[0];
  const float* revin_sc   = (const float*)d_in[1];
  const float* revin_bi   = (const float*)d_in[2];
  const float* up_w       = (const float*)d_in[3];
  const float* lstm_wi    = (const float*)d_in[4];
  const float* lstm_wh    = (const float*)d_in[5];
  const float* lstm_b     = (const float*)d_in[6];
  const float* norm_scale = (const float*)d_in[7];
  const float* wq         = (const float*)d_in[8];
  const float* wk         = (const float*)d_in[9];
  const float* wv         = (const float*)d_in[10];
  const float* wo         = (const float*)d_in[11];
  const float* down_w     = (const float*)d_in[12];
  const float* head_ns    = (const float*)d_in[13];
  const float* point_w    = (const float*)d_in[14];
  const float* quant_w    = (const float*)d_in[15];
  float* out = (float*)d_out;

  // Workspace carve-out (256B aligned).
  size_t off = 0;
  auto alloc = [&](size_t bytes) -> void* {
    void* p = (char*)d_ws + off;
    off += (bytes + 255) & ~(size_t)255;
    return p;
  };
  const size_t NR = 64512;  // b*m*n_p*d rows
  float* mean   = (float*)alloc(1024 * 4);
  float* stdv   = (float*)alloc(1024 * 4);
  float* invstd = (float*)alloc(1024 * 4);
  unsigned short* WlstmT = (unsigned short*)alloc((size_t)4 * 2048 * 768 * 2);
  unsigned short* WqT    = (unsigned short*)alloc((size_t)4 * 512 * 512 * 2);
  unsigned short* WkT    = (unsigned short*)alloc((size_t)4 * 128 * 512 * 2);
  unsigned short* WvT    = (unsigned short*)alloc((size_t)4 * 128 * 512 * 2);
  unsigned short* WoT    = (unsigned short*)alloc((size_t)4 * 512 * 512 * 2);
  unsigned short* WdownT = (unsigned short*)alloc((size_t)4 * 256 * 512 * 2);
  unsigned short* pointT = (unsigned short*)alloc((size_t)96 * 256 * 2);
  unsigned short* quantT = (unsigned short*)alloc((size_t)288 * 256 * 2);
  unsigned short* Xb   = (unsigned short*)alloc((size_t)1024 * 63 * 256 * 2);
  unsigned short* Hbuf = (unsigned short*)alloc((size_t)2 * 1024 * 512 * 2);
  float*          Cbuf = (float*)alloc((size_t)1024 * 512 * 4);
  unsigned short* HS   = (unsigned short*)alloc(NR * 512 * 2);
  unsigned short* NRM  = (unsigned short*)alloc(NR * 512 * 2);
  unsigned short* Qb   = (unsigned short*)alloc(NR * 512 * 2);
  unsigned short* Kb   = (unsigned short*)alloc(NR * 128 * 2);
  unsigned short* Vb   = (unsigned short*)alloc(NR * 128 * 2);
  unsigned short* NBuf = (unsigned short*)alloc((size_t)1024 * 256 * 2);
  float* POINT = (float*)alloc((size_t)1024 * 96 * 4);
  float* QUANT = (float*)alloc((size_t)1024 * 288 * 4);
  // Aliased buffers (lifetimes do not overlap within a block iteration).
  unsigned short* AO  = HS;   // attention out reuses HS (dead after rmsnorm)
  unsigned short* MIX = NRM;  // wo output reuses NRM (dead after q/k/v gemms)
  unsigned short* BEL = Qb;   // down output reuses Qb (dead after attention)

  auto transpose = [&](const float* src, int K, int N, unsigned short* dst,
                       int dstStride, int colOff) {
    int total = K * N;
    transpose_bf16_kernel<<<(total + 255) / 256, 256, 0, stream>>>(
        src, K, N, dst, dstStride, colOff);
  };
  auto gemm = [&](const unsigned short* A, int lda, const unsigned short* Bt,
                  int ldb, int M, int N, int K, float* outF,
                  unsigned short* outB, int ldo) {
    dim3 grid(M / 64, (N + 63) / 64);
    gemm_bf16_kernel<<<grid, 256, 0, stream>>>(A, lda, Bt, ldb, N, K, outF,
                                               outB, ldo);
  };

  // ---- Stage 0: RevIN stats, patches, up-projection, weight transposes ----
  revin_stats_kernel<<<1024, 256, 0, stream>>>(series, mean, stdv, invstd);
  patch_up_kernel<<<1024 * 63, 256, 0, stream>>>(series, mean, invstd, revin_sc,
                                                 revin_bi, up_w, Xb);
  for (int blk = 0; blk < 4; blk++) {
    unsigned short* wl = WlstmT + (size_t)blk * 2048 * 768;
    transpose(lstm_wi + (size_t)blk * 256 * 2048, 256, 2048, wl, 768, 0);
    transpose(lstm_wh + (size_t)blk * 512 * 2048, 512, 2048, wl, 768, 256);
    transpose(wq + (size_t)blk * 512 * 512, 512, 512,
              WqT + (size_t)blk * 512 * 512, 512, 0);
    transpose(wk + (size_t)blk * 512 * 128, 512, 128,
              WkT + (size_t)blk * 128 * 512, 512, 0);
    transpose(wv + (size_t)blk * 512 * 128, 512, 128,
              WvT + (size_t)blk * 128 * 512, 512, 0);
    transpose(wo + (size_t)blk * 512 * 512, 512, 512,
              WoT + (size_t)blk * 512 * 512, 512, 0);
    transpose(down_w + (size_t)blk * 512 * 256, 512, 256,
              WdownT + (size_t)blk * 256 * 512, 512, 0);
  }
  transpose(point_w, 256, 96, pointT, 256, 0);
  transpose(quant_w, 256, 288, quantT, 256, 0);

  // ---- Stage 1..4: blocks ----
  for (int blk = 0; blk < 4; blk++) {
    hipMemsetAsync(Hbuf, 0, (size_t)1024 * 512 * 2, stream);  // h_0 = 0
    hipMemsetAsync(Cbuf, 0, (size_t)1024 * 512 * 4, stream);  // c_0 = 0
    const unsigned short* wl = WlstmT + (size_t)blk * 2048 * 768;
    const float* bb = lstm_b + (size_t)blk * 2048;
    for (int t = 0; t < 63; t++) {
      unsigned short* hp = Hbuf + (size_t)(t & 1) * 1024 * 512;
      unsigned short* hn = Hbuf + (size_t)((t + 1) & 1) * 1024 * 512;
      lstm_step_kernel<<<dim3(16, 8), 256, 0, stream>>>(Xb, hp, hn, Cbuf, wl,
                                                        bb, HS, t);
    }
    rmsnorm_kernel<<<(int)NR, 256, 0, stream>>>(HS, 512, 512,
                                                norm_scale + blk * 512, NRM, 512);
    gemm(NRM, 512, WqT + (size_t)blk * 512 * 512, 512, (int)NR, 512, 512,
         nullptr, Qb, 512);
    gemm(NRM, 512, WkT + (size_t)blk * 128 * 512, 512, (int)NR, 128, 512,
         nullptr, Kb, 128);
    gemm(NRM, 512, WvT + (size_t)blk * 128 * 512, 512, (int)NR, 128, 512,
         nullptr, Vb, 128);
    attn_kernel<<<4032 * 8, 256, 0, stream>>>(Qb, Kb, Vb, AO);
    gemm(AO, 512, WoT + (size_t)blk * 512 * 512, 512, (int)NR, 512, 512,
         nullptr, MIX, 512);
    gemm(MIX, 512, WdownT + (size_t)blk * 256 * 512, 512, (int)NR, 256, 512,
         nullptr, BEL, 256);
    permute_kernel<<<(int)NR, 256, 0, stream>>>(BEL, Xb);
  }

  // ---- Stage 5: head ----
  rmsnorm_kernel<<<1024, 256, 0, stream>>>(Xb + 62 * 256, 63 * 256, 256,
                                           head_ns, NBuf, 256);
  gemm(NBuf, 256, pointT, 256, 1024, 96, 256, POINT, nullptr, 96);
  gemm(NBuf, 256, quantT, 256, 1024, 288, 256, QUANT, nullptr, 288);
  head_final_kernel<<<(4 * 1024 * 96) / 256, 256, 0, stream>>>(
      POINT, QUANT, revin_sc, revin_bi, mean, stdv, out);
}
